// MultiHeadAttention_5214090297711
// MI455X (gfx1250) — matrix operand order repaired and checked by reading, not compiled, not run on hardware
//
#include <hip/hip_runtime.h>
#include <stddef.h>

// ---------------------------------------------------------------------------
// MultiHeadAttention forward for MI455X (gfx1250, wave32, WMMA bf16)
//   B=4, L=2048, D_MODEL=1024, H=16, D_K=64
//   Pipeline:
//     1) cast fp32 inputs/weights -> bf16
//     2) Q/K/V projections: LDS-tiled, register-double-buffered WMMA GEMM,
//        scatter to [B,H,L,64] bf16 (Q scaled by 1/sqrt(64))
//     3) flash attention (online softmax) in registers/LDS -> Oh bf16
//     4) output projection: WMMA GEMM -> fp32 d_out
// ---------------------------------------------------------------------------

#define BATCH 4
#define SEQ   2048
#define DM    1024
#define NHEAD 16
#define DK    64
#define MROWS (BATCH * SEQ)          // 8192
#define NEGBIG (-1000000000.0f)

typedef __attribute__((ext_vector_type(16))) __bf16 bf16x16;
typedef __attribute__((ext_vector_type(8)))  __bf16 bf16x8;
typedef __attribute__((ext_vector_type(8)))  float  f32x8;

// A-operand (16x32 bf16): lane = (row = lane&15, hi = lane>>4)
//   vector elems 0..7  <-> K = hi*16 + e
//   vector elems 8..15 <-> K = hi*16 + 8 + (e-8)
__device__ __forceinline__ bf16x16 ld_a16(const __bf16* p_lo, const __bf16* p_hi) {
  bf16x8 lo = *(const bf16x8*)p_lo;
  bf16x8 hi = *(const bf16x8*)p_hi;
  bf16x16 r;
#pragma unroll
  for (int i = 0; i < 8; ++i) { r[i] = lo[i]; r[i + 8] = hi[i]; }
  return r;
}

// B-operand (32x16 bf16): lane = (col = lane&15, hi = lane>>4)
//   vector elems 0..15 <-> K = hi*16 + e  (contiguous 32 bytes)
__device__ __forceinline__ bf16x16 ld_b16(const __bf16* p) {
  return *(const bf16x16*)p;
}

__device__ __forceinline__ f32x8 wmma_bf16(bf16x16 a, bf16x16 b, f32x8 c) {
  return __builtin_amdgcn_wmma_f32_16x16x32_bf16(
      /*neg_a=*/false, a, /*neg_b=*/false, b,
      /*c_mod=*/(short)0, c, /*reuse_a=*/false, /*reuse_b=*/false);
}

// ---------------------------------------------------------------------------
// 1) fp32 -> bf16 cast
// ---------------------------------------------------------------------------
__global__ void cast_f32_to_bf16(const float* __restrict__ src,
                                 __bf16* __restrict__ dst, int n) {
  int i = blockIdx.x * blockDim.x + threadIdx.x;
  int stride = gridDim.x * blockDim.x;
  for (; i < n; i += stride) dst[i] = (__bf16)src[i];
}

// ---------------------------------------------------------------------------
// 2/4) GEMM: C[m,n] = scale * sum_k A[m,k] * W[n,k]   (x @ W^T)
//   M=8192, N=1024, K=1024.  Block tile 128(M) x 128(N), 256 thr = 8 waves,
//   wave tile 32x64 (2x4 WMMA frags), k-step 32 staged through LDS with
//   register double-buffering of the global loads (no EXEC divergence).
//   outH != nullptr : scatter bf16 into head-major [B,H,L,DK]
//   outH == nullptr : write fp32 row-major [M, DM] to outF
// ---------------------------------------------------------------------------
__global__ __launch_bounds__(256)
void gemm_xwT_wmma(const __bf16* __restrict__ A, const __bf16* __restrict__ W,
                   __bf16* __restrict__ outH, float* __restrict__ outF,
                   float scale) {
  __shared__ __bf16 As[128 * 32];   // 8 KB
  __shared__ __bf16 Bs[128 * 32];   // 8 KB

  const int lane = threadIdx.x & 31;
  const int wid  = threadIdx.x >> 5;      // 0..7
  const int wm   = wid >> 1;              // 0..3  (M sub-tile, 32 rows)
  const int wn   = wid & 1;               // 0..1  (N sub-tile, 64 cols)
  const int lr   = lane & 15;
  const int hi   = lane >> 4;
  const int m0   = blockIdx.y * 128;
  const int n0   = blockIdx.x * 128;

  // cooperative staging slot: every thread moves 16 elems of A and 16 of W
  const int srow = threadIdx.x >> 1;          // 0..127
  const int scg  = (threadIdx.x & 1) * 16;    // 0 / 16
  const __bf16* aptr = A + (size_t)(m0 + srow) * DM + scg;
  const __bf16* wptr = W + (size_t)(n0 + srow) * DM + scg;

  // register double buffer (k-step 0 preloaded)
  bf16x8 ra0 = *(const bf16x8*)aptr;
  bf16x8 ra1 = *(const bf16x8*)(aptr + 8);
  bf16x8 rb0 = *(const bf16x8*)wptr;
  bf16x8 rb1 = *(const bf16x8*)(wptr + 8);

  f32x8 c[2][4];
#pragma unroll
  for (int i = 0; i < 2; ++i)
#pragma unroll
    for (int j = 0; j < 4; ++j)
      c[i][j] = (f32x8){0.f, 0.f, 0.f, 0.f, 0.f, 0.f, 0.f, 0.f};

  for (int k0 = 0; k0 < DM; k0 += 32) {
    // stage current k-slab into LDS
    *(bf16x8*)&As[srow * 32 + scg]     = ra0;
    *(bf16x8*)&As[srow * 32 + scg + 8] = ra1;
    *(bf16x8*)&Bs[srow * 32 + scg]     = rb0;
    *(bf16x8*)&Bs[srow * 32 + scg + 8] = rb1;
    __syncthreads();

    // issue next slab's global loads; they retire under the WMMAs below
    if (k0 + 32 < DM) {
      ra0 = *(const bf16x8*)(aptr + k0 + 32);
      ra1 = *(const bf16x8*)(aptr + k0 + 40);
      rb0 = *(const bf16x8*)(wptr + k0 + 32);
      rb1 = *(const bf16x8*)(wptr + k0 + 40);
    }

    bf16x16 a[2], b[4];
#pragma unroll
    for (int i = 0; i < 2; ++i) {
      const __bf16* row = &As[(wm * 32 + i * 16 + lr) * 32];
      a[i] = ld_a16(row + hi * 16, row + hi * 16 + 8);
    }
#pragma unroll
    for (int j = 0; j < 4; ++j)
      b[j] = ld_b16(&Bs[(wn * 64 + j * 16 + lr) * 32 + hi * 16]);

#pragma unroll
    for (int i = 0; i < 2; ++i)
#pragma unroll
      for (int j = 0; j < 4; ++j)
        c[i][j] = wmma_bf16(a[i], b[j], c[i][j]);
    __syncthreads();
  }

  // epilogue: C layout -> VGPR r: (M = r + hi*8, N = lr)
#pragma unroll
  for (int i = 0; i < 2; ++i)
#pragma unroll
    for (int j = 0; j < 4; ++j)
#pragma unroll
      for (int r = 0; r < 8; ++r) {
        int gr = m0 + wm * 32 + i * 16 + r + hi * 8;
        int gc = n0 + wn * 64 + j * 16 + lr;
        float v = c[i][j][r] * scale;
        if (outH) {
          int bb = gr >> 11, ll = gr & (SEQ - 1);
          int hh = gc >> 6,  dd = gc & (DK - 1);
          outH[(((size_t)bb * NHEAD + hh) * SEQ + ll) * DK + dd] = (__bf16)v;
        } else {
          outF[(size_t)gr * DM + gc] = v;
        }
      }
}

// ---------------------------------------------------------------------------
// 3) Flash attention.  Grid: (L/64, H, B); block 128 thr = 4 waves; each wave
//    owns a 16-row Q block. Keys/values streamed in 32-wide chunks via LDS,
//    register-double-buffered so global loads overlap the WMMAs/softmax.
// ---------------------------------------------------------------------------
__global__ __launch_bounds__(128)
void flash_attn_wmma(const __bf16* __restrict__ Qh, const __bf16* __restrict__ Kh,
                     const __bf16* __restrict__ Vh, const int* __restrict__ mask,
                     __bf16* __restrict__ Oh) {
  __shared__ __bf16 sK[32 * 64];       // key chunk, row-major [key][d]   4 KB
  __shared__ __bf16 sVt[64 * 32];      // value chunk transposed [d][key] 4 KB
  __shared__ __bf16 sP[4][16 * 32];    // per-wave P scratch              4 KB

  const int lane = threadIdx.x & 31;
  const int wid  = threadIdx.x >> 5;   // 0..3
  const int lr   = lane & 15;
  const int hi   = lane >> 4;
  const int bb   = blockIdx.z;
  const int hh   = blockIdx.y;
  const int q0   = blockIdx.x * 64 + wid * 16;

  const __bf16* Qb = Qh + ((size_t)(bb * NHEAD + hh) * SEQ) * DK;
  const __bf16* Kb = Kh + ((size_t)(bb * NHEAD + hh) * SEQ) * DK;
  const __bf16* Vb = Vh + ((size_t)(bb * NHEAD + hh) * SEQ) * DK;
  const int*    Mb = mask + (size_t)bb * SEQ * SEQ;

  // Q A-frags for K-dim 0..31 and 32..63 (Q already scaled by 1/sqrt(dk))
  const __bf16* qrow = Qb + (size_t)(q0 + lr) * DK;
  bf16x16 aq0 = ld_a16(qrow + hi * 16,      qrow + hi * 16 + 8);
  bf16x16 aq1 = ld_a16(qrow + 32 + hi * 16, qrow + 40 + hi * 16);

  // cooperative K/V staging slot: thread -> (key row 0..31, 16-wide d segment)
  const int crow = threadIdx.x >> 2;         // 0..31
  const int cseg = (threadIdx.x & 3) * 16;   // 0/16/32/48
  const __bf16* kbase = Kb + (size_t)crow * DK + cseg;
  const __bf16* vbase = Vb + (size_t)crow * DK + cseg;

  // register double buffer (chunk 0 preloaded)
  bf16x8 rk0 = *(const bf16x8*)kbase;
  bf16x8 rk1 = *(const bf16x8*)(kbase + 8);
  bf16x8 rv0 = *(const bf16x8*)vbase;
  bf16x8 rv1 = *(const bf16x8*)(vbase + 8);

  f32x8 oacc[4];
#pragma unroll
  for (int j = 0; j < 4; ++j)
    oacc[j] = (f32x8){0.f, 0.f, 0.f, 0.f, 0.f, 0.f, 0.f, 0.f};
  float rm[8], rl[8];
#pragma unroll
  for (int r = 0; r < 8; ++r) { rm[r] = -3.0e38f; rl[r] = 0.f; }

  for (int kc = 0; kc < SEQ; kc += 32) {
    // stage current chunk: K row-major, V transposed
    *(bf16x8*)&sK[crow * 64 + cseg]     = rk0;
    *(bf16x8*)&sK[crow * 64 + cseg + 8] = rk1;
#pragma unroll
    for (int i = 0; i < 8; ++i) {
      sVt[(cseg + i) * 32 + crow]     = rv0[i];
      sVt[(cseg + 8 + i) * 32 + crow] = rv1[i];
    }
    __syncthreads();

    // issue next chunk's global loads; they retire under the math below
    if (kc + 32 < SEQ) {
      size_t off = (size_t)(kc + 32) * DK;
      rk0 = *(const bf16x8*)(kbase + off);
      rk1 = *(const bf16x8*)(kbase + off + 8);
      rv0 = *(const bf16x8*)(vbase + off);
      rv1 = *(const bf16x8*)(vbase + off + 8);
    }

    // ---- S = Q @ K^T for this 16x32 key strip (2 score frags) ----
    f32x8 s0 = (f32x8){0.f, 0.f, 0.f, 0.f, 0.f, 0.f, 0.f, 0.f};
    f32x8 s1 = s0;
    {
      const __bf16* k0r = &sK[lr * 64];          // keys kc+0..15
      const __bf16* k1r = &sK[(16 + lr) * 64];   // keys kc+16..31
      s0 = wmma_bf16(aq0, ld_b16(k0r + hi * 16), s0);
      s0 = wmma_bf16(aq1, ld_b16(k0r + 32 + hi * 16), s0);
      s1 = wmma_bf16(aq0, ld_b16(k1r + hi * 16), s1);
      s1 = wmma_bf16(aq1, ld_b16(k1r + 32 + hi * 16), s1);
    }

    // ---- mask ----
#pragma unroll
    for (int r = 0; r < 8; ++r) {
      int row = q0 + r + hi * 8;
      const int* mrow = Mb + (size_t)row * SEQ + kc;
      if (mrow[lr] == 0)      s0[r] = NEGBIG;
      if (mrow[16 + lr] == 0) s1[r] = NEGBIG;
    }

    // ---- online softmax update (rows live in (VGPR r, half) pairs) ----
#pragma unroll
    for (int r = 0; r < 8; ++r) {
      float mx = fmaxf(s0[r], s1[r]);
      mx = fmaxf(mx, __shfl_xor(mx, 1));
      mx = fmaxf(mx, __shfl_xor(mx, 2));
      mx = fmaxf(mx, __shfl_xor(mx, 4));
      mx = fmaxf(mx, __shfl_xor(mx, 8));
      float mnew = fmaxf(rm[r], mx);
      float corr = __expf(rm[r] - mnew);
      rm[r] = mnew;
      float p0 = __expf(s0[r] - mnew);
      float p1 = __expf(s1[r] - mnew);
      s0[r] = p0; s1[r] = p1;
      float ps = p0 + p1;
      ps += __shfl_xor(ps, 1);
      ps += __shfl_xor(ps, 2);
      ps += __shfl_xor(ps, 4);
      ps += __shfl_xor(ps, 8);
      rl[r] = rl[r] * corr + ps;
      oacc[0][r] *= corr; oacc[1][r] *= corr;
      oacc[2][r] *= corr; oacc[3][r] *= corr;
    }

    // ---- P: C-layout -> A-layout via per-wave LDS scratch ----
    __bf16* pp = &sP[wid][0];
#pragma unroll
    for (int r = 0; r < 8; ++r) {
      int row = r + hi * 8;
      pp[row * 32 + lr]      = (__bf16)s0[r];
      pp[row * 32 + 16 + lr] = (__bf16)s1[r];
    }
    bf16x16 pa = ld_a16(&pp[lr * 32 + hi * 16], &pp[lr * 32 + hi * 16 + 8]);

    // ---- O += P @ V (4 d-strips of 16) ----
#pragma unroll
    for (int j = 0; j < 4; ++j) {
      bf16x16 bv = ld_b16(&sVt[(j * 16 + lr) * 32 + hi * 16]);
      oacc[j] = wmma_bf16(pa, bv, oacc[j]);
    }
    __syncthreads();
  }

  // ---- finalize: divide by row sums, store bf16 head-interleaved [B*L, DM] ----
  float inv[8];
#pragma unroll
  for (int r = 0; r < 8; ++r) inv[r] = 1.0f / rl[r];
#pragma unroll
  for (int j = 0; j < 4; ++j)
#pragma unroll
    for (int r = 0; r < 8; ++r) {
      int row = q0 + r + hi * 8;
      int col = hh * DK + j * 16 + lr;
      Oh[((size_t)bb * SEQ + row) * DM + col] = (__bf16)(oacc[j][r] * inv[r]);
    }
}

// ---------------------------------------------------------------------------
// launcher
// ---------------------------------------------------------------------------
extern "C" void kernel_launch(void* const* d_in, const int* in_sizes, int n_in,
                              void* d_out, int out_size, void* d_ws, size_t ws_size,
                              hipStream_t stream) {
  const float* q   = (const float*)d_in[0];
  const float* k   = (const float*)d_in[1];
  const float* v   = (const float*)d_in[2];
  const float* w_q = (const float*)d_in[3];
  const float* w_k = (const float*)d_in[4];
  const float* w_v = (const float*)d_in[5];
  const float* w_o = (const float*)d_in[6];
  const int*  mask = (const int*)d_in[7];
  float* out = (float*)d_out;

  const size_t MB = 1024ull * 1024ull;
  char* ws = (char*)d_ws;
  __bf16* qb  = (__bf16*)(ws + 0 * MB);     // 16 MB  [8192,1024]
  __bf16* kb  = (__bf16*)(ws + 16 * MB);    // 16 MB
  __bf16* vb  = (__bf16*)(ws + 32 * MB);    // 16 MB
  __bf16* wqb = (__bf16*)(ws + 48 * MB);    //  2 MB  [1024,1024]
  __bf16* wkb = (__bf16*)(ws + 50 * MB);
  __bf16* wvb = (__bf16*)(ws + 52 * MB);
  __bf16* wob = (__bf16*)(ws + 54 * MB);
  __bf16* Qh  = (__bf16*)(ws + 56 * MB);    // 16 MB  [B,H,L,DK] (pre-scaled)
  __bf16* Kh  = (__bf16*)(ws + 72 * MB);
  __bf16* Vh  = (__bf16*)(ws + 88 * MB);
  __bf16* Oh  = (__bf16*)(ws + 104 * MB);   // 16 MB  [B*L, DM]

  const int nAct = MROWS * DM;   // 8,388,608
  const int nW   = DM * DM;      // 1,048,576
  cast_f32_to_bf16<<<2048, 256, 0, stream>>>(q,   qb,  nAct);
  cast_f32_to_bf16<<<2048, 256, 0, stream>>>(k,   kb,  nAct);
  cast_f32_to_bf16<<<2048, 256, 0, stream>>>(v,   vb,  nAct);
  cast_f32_to_bf16<<<512,  256, 0, stream>>>(w_q, wqb, nW);
  cast_f32_to_bf16<<<512,  256, 0, stream>>>(w_k, wkb, nW);
  cast_f32_to_bf16<<<512,  256, 0, stream>>>(w_v, wvb, nW);
  cast_f32_to_bf16<<<512,  256, 0, stream>>>(w_o, wob, nW);

  dim3 ggrid(DM / 128, MROWS / 128);  // (8, 64)
  const float qscale = 0.125f;        // 1/sqrt(DK)
  gemm_xwT_wmma<<<ggrid, 256, 0, stream>>>(qb, wqb, Qh, nullptr, qscale);
  gemm_xwT_wmma<<<ggrid, 256, 0, stream>>>(kb, wkb, Kh, nullptr, 1.0f);
  gemm_xwT_wmma<<<ggrid, 256, 0, stream>>>(vb, wvb, Vh, nullptr, 1.0f);

  dim3 agrid(SEQ / 64, NHEAD, BATCH); // (32, 16, 4)
  flash_attn_wmma<<<agrid, 128, 0, stream>>>(Qh, Kh, Vh, mask, Oh);

  gemm_xwT_wmma<<<ggrid, 256, 0, stream>>>(Oh, wob, nullptr, out, 1.0f);
}
